// Block_16853451670038
// MI455X (gfx1250) — matrix operand order repaired and checked by reading, not compiled, not run on hardware
//
#include <hip/hip_runtime.h>
#include <hip/hip_bf16.h>
#include <math.h>

// ---------------------------------------------------------------- types
typedef __bf16 bf16;
typedef __attribute__((ext_vector_type(16))) __bf16 v16bf;
typedef __attribute__((ext_vector_type(8)))  float  v8f;

#define B_   4
#define T_   2048
#define C_   1024
#define H_   128
#define E_   8
#define I_   1024
#define NTOK (B_ * T_)
#define LSTR 40            // LDS tile row stride in bf16 elems (80B, 16B multiple)
#define NEG_BIG (-1.0e30f)

__device__ __forceinline__ bf16 f2bf(float x) { return (bf16)x; }

__device__ __forceinline__ v8f wmma_bf16(v16bf a, v16bf b, v8f c) {
  // D = A(16x32 bf16) * B(32x16 bf16) + C(16x16 f32)
  return __builtin_amdgcn_wmma_f32_16x16x32_bf16(false, a, false, b, (short)0, c,
                                                 false, false);
}

// ---------------------------------------------------------------- tile loaders
// Copy `rows` x 32 (row-major source) into LDS [rows][LSTR]. 256 threads.
__device__ __forceinline__ void load_tile_a(bf16* __restrict__ lds,
                                            const bf16* __restrict__ src,
                                            long row0, int col0, int ld,
                                            int rows) {
  for (int idx = threadIdx.x; idx < rows * 4; idx += 256) {
    int r = idx >> 2;
    int seg = idx & 3;
    const bf16* s = src + (row0 + r) * (long)ld + col0 + seg * 8;
    uint4 v = *(const uint4*)s;
    *(uint4*)(lds + r * LSTR + seg * 8) = v;
  }
}

// A-tile staging with per-row scaling: lds[r][k] = src[r][k] * scale[r].
// Used to fold the expert routing weight into A (diag(rw_e) * A @ W_e).
__device__ __forceinline__ void load_tile_a_scaled(bf16* __restrict__ lds,
                                                   const bf16* __restrict__ src,
                                                   long row0, int col0, int ld,
                                                   int rows,
                                                   const float* __restrict__ scl,
                                                   int e) {
  for (int idx = threadIdx.x; idx < rows * 4; idx += 256) {
    int r = idx >> 2;
    int seg = idx & 3;
    const bf16* s = src + (row0 + r) * (long)ld + col0 + seg * 8;
    uint4 v = *(const uint4*)s;
    const bf16* in = (const bf16*)&v;
    float f = scl[r * E_ + e];
    unsigned short outp[8];
#pragma unroll
    for (int j = 0; j < 8; ++j) {
      bf16 b = f2bf((float)in[j] * f);
      outp[j] = *(const unsigned short*)&b;
    }
    *(uint4*)(lds + r * LSTR + seg * 8) = *(const uint4*)outp;
  }
}

// Same copy but through the async Global->LDS DMA path (ASYNCcnt-tracked);
// caller must issue s_wait_asynccnt before the block barrier.
__device__ __forceinline__ void load_tile_a_async(bf16* __restrict__ lds,
                                                  const bf16* __restrict__ src,
                                                  long row0, int col0, int ld,
                                                  int rows) {
  for (int idx = threadIdx.x; idx < rows * 4; idx += 256) {
    int r = idx >> 2;
    int seg = idx & 3;
    const bf16* s = src + (row0 + r) * (long)ld + col0 + seg * 8;
    bf16* d = lds + r * LSTR + seg * 8;
    unsigned dst_off =
        (unsigned)(size_t)(__attribute__((address_space(3))) bf16*)d;
    unsigned long long ga = (unsigned long long)(size_t)s;
    asm volatile("global_load_async_to_lds_b128 %0, %1, off"
                 :
                 : "v"(dst_off), "v"(ga)
                 : "memory");
  }
}
__device__ __forceinline__ void wait_async_lds() {
  asm volatile("s_wait_asynccnt 0x0" ::: "memory");
}

// Source is [K][N] row-major; fill LDS Bs[n][k] (64 n x 32 k) with K-pairs
// packed in registers so LDS stores are b32 (half the store count of b16).
__device__ __forceinline__ void load_tile_b(bf16* __restrict__ lds,
                                            const bf16* __restrict__ src,
                                            long k0, int n0, int ld) {
  int t = threadIdx.x;
  int kp = t >> 4;         // 0..15 -> k pair (2kp, 2kp+1)
  int nseg = t & 15;       // 0..15 -> n chunk of 4
  const bf16* s0 = src + (k0 + 2 * kp) * (long)ld + n0 + nseg * 4;
  const bf16* s1 = s0 + ld;
  uint2 r0 = *(const uint2*)s0;
  uint2 r1 = *(const uint2*)s1;
  const unsigned short* a0 = (const unsigned short*)&r0;
  const unsigned short* a1 = (const unsigned short*)&r1;
#pragma unroll
  for (int j = 0; j < 4; ++j) {
    unsigned pk = (unsigned)a0[j] | ((unsigned)a1[j] << 16);
    *(unsigned*)(lds + (nseg * 4 + j) * LSTR + 2 * kp) = pk;
  }
}

// ---------------------------------------------------------------- fragments
// A fragment (16x32): lane 0-15 rows M=lane (K 0..7,16..23), lane 16-31 same
// rows, K 8..15,24..31; VGPR v holds a K-pair (ISA 7.12.2, 16-bit A layout).
__device__ __forceinline__ v16bf frag_A(const bf16* __restrict__ As, int mt) {
  int lane = threadIdx.x & 31;
  int hl = lane >> 4, r = lane & 15;
  const bf16* p = As + (mt * 16 + r) * LSTR;
  v16bf a;
  unsigned int* au = (unsigned int*)&a;
#pragma unroll
  for (int v = 0; v < 4; ++v) au[v] = *(const unsigned int*)(p + hl * 8 + 2 * v);
#pragma unroll
  for (int v = 4; v < 8; ++v)
    au[v] = *(const unsigned int*)(p + 16 + hl * 8 + 2 * (v - 4));
  return a;
}

// B fragment (32x16): lane 0-15 col N=lane K 0..7,16..23, lane 16-31 K 8..15,24..31.
__device__ __forceinline__ v16bf frag_B(const bf16* __restrict__ Bs, int nt) {
  int lane = threadIdx.x & 31;
  int hl = lane >> 4, n = lane & 15;
  const bf16* p = Bs + (nt * 16 + n) * LSTR;
  v16bf b;
  unsigned int* bu = (unsigned int*)&b;
#pragma unroll
  for (int v = 0; v < 8; ++v)
    bu[v] = *(const unsigned int*)(p + (v >> 2) * 16 + hl * 8 + 2 * (v & 3));
  return b;
}

// ---------------------------------------------------------------- pack kernels
__global__ __launch_bounds__(256) void k_pack_qkv(const float* __restrict__ q,
                                                  const float* __restrict__ k,
                                                  const float* __restrict__ v,
                                                  bf16* __restrict__ wqkv) {
  long i = (long)blockIdx.x * 256 + threadIdx.x;
  if (i >= (long)E_ * C_ * H_) return;
  int h = (int)(i % H_);
  long ec = i / H_;                       // e*C + c
  long src = ec * H_ + h;
  long dst = ec * (3 * H_);
  wqkv[dst + h]          = f2bf(q[src]);
  wqkv[dst + H_ + h]     = f2bf(k[src]);
  wqkv[dst + 2 * H_ + h] = f2bf(v[src]);
}

__global__ __launch_bounds__(256) void k_cvt_bf16(const float* __restrict__ s,
                                                  bf16* __restrict__ d, long n) {
  long i = (long)blockIdx.x * 256 + threadIdx.x;
  if (i < n) d[i] = f2bf(s[i]);
}

// ---------------------------------------------------------------- LN + gating
__global__ __launch_bounds__(256) void k_ln_gate(
    const float* __restrict__ xin, const float* __restrict__ lnw,
    const float* __restrict__ lnb, const float* __restrict__ sim,
    const float* __restrict__ gates, bf16* __restrict__ nout,
    float* __restrict__ rw, float* __restrict__ copy_out) {
  const long tok = blockIdx.x;
  const int tid = threadIdx.x;
  const int lane = tid & 31, wave = tid >> 5;
  const float* xr = xin + tok * C_;

  __shared__ float red[9][18];

  float x[4];
#pragma unroll
  for (int i = 0; i < 4; ++i) x[i] = xr[tid + i * 256];
  if (copy_out) {
    float* o = copy_out + tok * C_;
#pragma unroll
    for (int i = 0; i < 4; ++i) o[tid + i * 256] = x[i];
  }

  float s = 0.f, sq = 0.f;
#pragma unroll
  for (int i = 0; i < 4; ++i) { s += x[i]; sq += x[i] * x[i]; }
#pragma unroll
  for (int o = 16; o > 0; o >>= 1) {
    s += __shfl_down(s, o, 32);
    sq += __shfl_down(sq, o, 32);
  }
  if (lane == 0) { red[wave][0] = s; red[wave][1] = sq; }
  __syncthreads();
  float ts = 0.f, tsq = 0.f;
#pragma unroll
  for (int w = 0; w < 8; ++w) { ts += red[w][0]; tsq += red[w][1]; }
  const float mean = ts * (1.0f / C_);
  const float var = tsq * (1.0f / C_) - mean * mean;
  const float rstd = rsqrtf(var + 1e-5f);
  __syncthreads();

  float vals[17];
#pragma unroll
  for (int i = 0; i < 17; ++i) vals[i] = 0.f;
#pragma unroll
  for (int i = 0; i < 4; ++i) {
    int c = tid + i * 256;
    float nv = (x[i] - mean) * rstd * lnw[c] + lnb[c];
    nout[tok * C_ + c] = f2bf(nv);
    vals[0] += nv * nv;
    const float* sr = sim + (long)c * E_;
#pragma unroll
    for (int e = 0; e < 8; ++e) {
      float sv = sr[e];
      vals[1 + e] += nv * sv;
      vals[9 + e] += sv * sv;
    }
  }
#pragma unroll
  for (int i = 0; i < 17; ++i) {
    float t = vals[i];
#pragma unroll
    for (int o = 16; o > 0; o >>= 1) t += __shfl_down(t, o, 32);
    if (lane == 0) red[wave][i] = t;
  }
  __syncthreads();
  if (tid == 0) {
    float tot[17];
#pragma unroll
    for (int i = 0; i < 17; ++i) {
      float a = 0.f;
#pragma unroll
      for (int w = 0; w < 8; ++w) a += red[w][i];
      tot[i] = a;
    }
    float dn = fmaxf(sqrtf(tot[0]), 1e-12f);
    float logit[8], gated[8];
    int mask[8], cnt = 0;
#pragma unroll
    for (int e = 0; e < 8; ++e) {
      float sn = fmaxf(sqrtf(tot[9 + e]), 1e-12f);
      float l = tot[1 + e] / (dn * sn) - 1.0f / (1.0f + __expf(-gates[e]));
      logit[e] = l;
      float g = fmaxf(l, 0.f);
      gated[e] = g;
      mask[e] = (g > 0.f) ? 1 : 0;
      cnt += mask[e];
    }
    if (cnt == 0) {  // top-2 fallback on raw logits
      int i1 = 0;
      for (int e = 1; e < 8; ++e) if (logit[e] > logit[i1]) i1 = e;
      int i2 = (i1 == 0) ? 1 : 0;
      for (int e = 0; e < 8; ++e)
        if (e != i1 && logit[e] > logit[i2]) i2 = e;
      mask[i1] = 1; mask[i2] = 1;
    }
    float mx = NEG_BIG;
#pragma unroll
    for (int e = 0; e < 8; ++e) if (mask[e]) mx = fmaxf(mx, gated[e]);
    float w[8], sm = 0.f;
#pragma unroll
    for (int e = 0; e < 8; ++e) {
      float ev = mask[e] ? __expf(gated[e] - mx) : 0.f;
      w[e] = ev; sm += ev;
    }
    float inv = 1.0f / sm;
#pragma unroll
    for (int e = 0; e < 8; ++e) rw[tok * E_ + e] = w[e] * inv;
  }
}

// ---------------------------------------------------------------- GEMM: rw-reduced
// D[m,n] = sum_e rw[m,e] * (A @ W_e)[m,n]  (+ residual), W is (E,K,N).
// The routing weight is folded into the A tile rows at staging time, so a
// single accumulator set is carried across experts (keeps VGPRs < 256).
__global__ __launch_bounds__(256) void k_gemm_rw(
    const bf16* __restrict__ A, int lda, const bf16* __restrict__ W, int K,
    int N, const float* __restrict__ rw, const float* __restrict__ residual,
    float* __restrict__ outF, bf16* __restrict__ outB, int ldo) {
  __shared__ __align__(16) bf16 As[128 * LSTR];
  __shared__ __align__(16) bf16 Bs[64 * LSTR];
  __shared__ float rwS[128 * E_];
  const int m0 = blockIdx.y * 128, n0 = blockIdx.x * 64;
  const int tid = threadIdx.x, lane = tid & 31, wave = tid >> 5;
  const int wmb = (wave & 3) * 2, wnb = (wave >> 2) * 2;
  for (int i = tid; i < 128 * E_; i += 256)
    rwS[i] = rw[(long)(m0 + (i >> 3)) * E_ + (i & 7)];

  v8f acc[2][2] = {{{}, {}}, {{}, {}}};
  for (int e = 0; e < E_; ++e) {
    const bf16* We = W + (long)e * K * N;
    for (int kk = 0; kk < K; kk += 32) {
      __syncthreads();
      load_tile_a_scaled(As, A, m0, kk, lda, 128, rwS, e);
      load_tile_b(Bs, We, kk, n0, N);
      if (kk + 32 < K)
        __builtin_prefetch(We + (long)(kk + 32) * N + n0 + (tid & 7) * 8, 0, 0);
      __syncthreads();
      v16bf a0 = frag_A(As, wmb), a1 = frag_A(As, wmb + 1);
      v16bf b0 = frag_B(Bs, wnb), b1 = frag_B(Bs, wnb + 1);
      acc[0][0] = wmma_bf16(a0, b0, acc[0][0]);
      acc[0][1] = wmma_bf16(a0, b1, acc[0][1]);
      acc[1][0] = wmma_bf16(a1, b0, acc[1][0]);
      acc[1][1] = wmma_bf16(a1, b1, acc[1][1]);
    }
  }
  const int hl = lane >> 4, c = lane & 15;
#pragma unroll
  for (int i = 0; i < 2; ++i) {
#pragma unroll
    for (int j = 0; j < 2; ++j) {
#pragma unroll
      for (int v = 0; v < 8; ++v) {
        long row = m0 + (wmb + i) * 16 + v + 8 * hl;
        int col = n0 + (wnb + j) * 16 + c;
        float val = acc[i][j][v];
        if (residual) val += residual[row * ldo + col];
        if (outF) outF[row * ldo + col] = val;
        if (outB) outB[row * ldo + col] = f2bf(val);
      }
    }
  }
}

// ---------------------------------------------------------------- attention
__global__ __launch_bounds__(256) void k_attn_scores(
    const bf16* __restrict__ qkv, float* __restrict__ scores, float scale) {
  const int b = blockIdx.z, t0 = blockIdx.y * 128, s0 = blockIdx.x * 64;
  if (s0 > t0 + 127) {  // fully above the diagonal: masked
    for (int i = threadIdx.x; i < 128 * 64; i += 256)
      scores[((long)b * T_ + t0 + (i >> 6)) * T_ + s0 + (i & 63)] = NEG_BIG;
    return;
  }
  __shared__ __align__(16) bf16 As[128 * LSTR];
  __shared__ __align__(16) bf16 Bs[64 * LSTR];
  const int lane = threadIdx.x & 31, wave = threadIdx.x >> 5;
  const int wmb = (wave & 3) * 2, wnb = (wave >> 2) * 2;
  const bf16* Q = qkv + (long)b * T_ * (3 * H_);         // [t][h]
  const bf16* Km = qkv + (long)b * T_ * (3 * H_) + H_;   // [s][h]
  v8f acc[2][2] = {{{}, {}}, {{}, {}}};
  for (int kk = 0; kk < H_; kk += 32) {
    __syncthreads();
    load_tile_a(As, Q, t0, kk, 3 * H_, 128);
    load_tile_a(Bs, Km, s0, kk, 3 * H_, 64);  // Bs[n=s][k=h]
    __syncthreads();
    v16bf a0 = frag_A(As, wmb), a1 = frag_A(As, wmb + 1);
    v16bf b0 = frag_B(Bs, wnb), b1 = frag_B(Bs, wnb + 1);
    acc[0][0] = wmma_bf16(a0, b0, acc[0][0]);
    acc[0][1] = wmma_bf16(a0, b1, acc[0][1]);
    acc[1][0] = wmma_bf16(a1, b0, acc[1][0]);
    acc[1][1] = wmma_bf16(a1, b1, acc[1][1]);
  }
  const int hl = lane >> 4, c = lane & 15;
#pragma unroll
  for (int i = 0; i < 2; ++i) {
#pragma unroll
    for (int j = 0; j < 2; ++j) {
#pragma unroll
      for (int v = 0; v < 8; ++v) {
        int trow = t0 + (wmb + i) * 16 + v + 8 * hl;
        int sc = s0 + (wnb + j) * 16 + c;
        float val = (sc > trow) ? NEG_BIG : acc[i][j][v] * scale;
        scores[((long)b * T_ + trow) * T_ + sc] = val;
      }
    }
  }
}

// Row softmax; writes bf16 probs in-place into the first half of each f32 row.
__global__ __launch_bounds__(256) void k_softmax(float* __restrict__ scores) {
  const long row = blockIdx.x;  // b*T + t
  float* sr = scores + row * T_;
  const int tid = threadIdx.x, lane = tid & 31, wave = tid >> 5;
  __shared__ float red[8];
  float v[8];
#pragma unroll
  for (int i = 0; i < 8; ++i) v[i] = sr[tid + i * 256];
  float mx = NEG_BIG;
#pragma unroll
  for (int i = 0; i < 8; ++i) mx = fmaxf(mx, v[i]);
#pragma unroll
  for (int o = 16; o > 0; o >>= 1) mx = fmaxf(mx, __shfl_down(mx, o, 32));
  if (lane == 0) red[wave] = mx;
  __syncthreads();
  mx = NEG_BIG;
#pragma unroll
  for (int w = 0; w < 8; ++w) mx = fmaxf(mx, red[w]);
  __syncthreads();
  float s = 0.f;
#pragma unroll
  for (int i = 0; i < 8; ++i) { v[i] = __expf(v[i] - mx); s += v[i]; }
#pragma unroll
  for (int o = 16; o > 0; o >>= 1) s += __shfl_down(s, o, 32);
  if (lane == 0) red[wave] = s;
  __syncthreads();
  s = 0.f;
#pragma unroll
  for (int w = 0; w < 8; ++w) s += red[w];
  float inv = 1.0f / s;
  __syncthreads();  // all reads of the f32 row done before bf16 overwrite
  bf16* pr = (bf16*)sr;
#pragma unroll
  for (int i = 0; i < 8; ++i) pr[tid + i * 256] = f2bf(v[i] * inv);
}

// attn = P @ V (per batch), causal K-loop truncation. P is bf16 in the score
// buffer with row stride 2*T_ elements.
__global__ __launch_bounds__(256) void k_gemm_pv(const float* __restrict__ scoresF,
                                                 const bf16* __restrict__ qkv,
                                                 bf16* __restrict__ attn) {
  __shared__ __align__(16) bf16 As[128 * LSTR];
  __shared__ __align__(16) bf16 Bs[64 * LSTR];
  const int b = blockIdx.z, t0 = blockIdx.y * 128, h0 = blockIdx.x * 64;
  const int lane = threadIdx.x & 31, wave = threadIdx.x >> 5;
  const int wmb = (wave & 3) * 2, wnb = (wave >> 2) * 2;
  const bf16* P = (const bf16*)scoresF;                    // row stride 2*T_
  const bf16* V = qkv + (long)b * T_ * (3 * H_) + 2 * H_;  // [s][h]
  v8f acc[2][2] = {{{}, {}}, {{}, {}}};
  const int kmax = t0 + 128;  // probs are zero for s > t
  for (int kk = 0; kk < kmax; kk += 32) {
    __syncthreads();
    load_tile_a(As, P, (long)b * T_ + t0, kk, 2 * T_, 128);
    load_tile_b(Bs, V, kk, h0, 3 * H_);
    __syncthreads();
    v16bf a0 = frag_A(As, wmb), a1 = frag_A(As, wmb + 1);
    v16bf b0 = frag_B(Bs, wnb), b1 = frag_B(Bs, wnb + 1);
    acc[0][0] = wmma_bf16(a0, b0, acc[0][0]);
    acc[0][1] = wmma_bf16(a0, b1, acc[0][1]);
    acc[1][0] = wmma_bf16(a1, b0, acc[1][0]);
    acc[1][1] = wmma_bf16(a1, b1, acc[1][1]);
  }
  const int hl = lane >> 4, c = lane & 15;
#pragma unroll
  for (int i = 0; i < 2; ++i) {
#pragma unroll
    for (int j = 0; j < 2; ++j) {
#pragma unroll
      for (int v = 0; v < 8; ++v) {
        long row = (long)b * T_ + t0 + (wmb + i) * 16 + v + 8 * hl;
        attn[row * H_ + h0 + (wnb + j) * 16 + c] = f2bf(acc[i][j][v]);
      }
    }
  }
}

// ---------------------------------------------------------------- MoE GEMMs
__global__ __launch_bounds__(256) void k_gemm_gelu(
    const bf16* __restrict__ A, const bf16* __restrict__ W1e,
    const float* __restrict__ rw2, int e, bf16* __restrict__ inter) {
  __shared__ __align__(16) bf16 As[128 * LSTR];
  __shared__ __align__(16) bf16 Bs[64 * LSTR];
  __shared__ float rwS[128];
  const int m0 = blockIdx.y * 128, n0 = blockIdx.x * 64;
  const int tid = threadIdx.x, lane = tid & 31, wave = tid >> 5;
  const int wmb = (wave & 3) * 2, wnb = (wave >> 2) * 2;
  if (tid < 128) rwS[tid] = rw2[(long)(m0 + tid) * E_ + e];
  v8f acc[2][2] = {{{}, {}}, {{}, {}}};
  for (int kk = 0; kk < C_; kk += 32) {
    __syncthreads();
    load_tile_a(As, A, m0, kk, C_, 128);
    load_tile_b(Bs, W1e, kk, n0, I_);
    if (kk + 32 < C_)
      __builtin_prefetch(W1e + (long)(kk + 32) * I_ + n0 + (tid & 7) * 8, 0, 0);
    __syncthreads();
    v16bf a0 = frag_A(As, wmb), a1 = frag_A(As, wmb + 1);
    v16bf b0 = frag_B(Bs, wnb), b1 = frag_B(Bs, wnb + 1);
    acc[0][0] = wmma_bf16(a0, b0, acc[0][0]);
    acc[0][1] = wmma_bf16(a0, b1, acc[0][1]);
    acc[1][0] = wmma_bf16(a1, b0, acc[1][0]);
    acc[1][1] = wmma_bf16(a1, b1, acc[1][1]);
  }
  const int hl = lane >> 4, c = lane & 15;
#pragma unroll
  for (int i = 0; i < 2; ++i) {
#pragma unroll
    for (int j = 0; j < 2; ++j) {
#pragma unroll
      for (int v = 0; v < 8; ++v) {
        int rl = (wmb + i) * 16 + v + 8 * hl;
        long row = m0 + rl;
        float f = rwS[rl];
        float xv = acc[i][j][v];
        float g = 0.5f * xv * (1.0f + erff(xv * 0.70710678f)) * f;
        inter[row * I_ + n0 + (wnb + j) * 16 + c] = f2bf(g);
      }
    }
  }
}

__global__ __launch_bounds__(256) void k_gemm_acc(const bf16* __restrict__ A,
                                                  const bf16* __restrict__ W2e,
                                                  float* __restrict__ out) {
  __shared__ __align__(16) bf16 As[128 * LSTR];
  __shared__ __align__(16) bf16 Bs[64 * LSTR];
  const int m0 = blockIdx.y * 128, n0 = blockIdx.x * 64;
  const int tid = threadIdx.x, lane = tid & 31, wave = tid >> 5;
  const int wmb = (wave & 3) * 2, wnb = (wave >> 2) * 2;
  v8f acc[2][2] = {{{}, {}}, {{}, {}}};
  for (int kk = 0; kk < I_; kk += 32) {
    __syncthreads();
    load_tile_a_async(As, A, m0, kk, I_, 128);   // ASYNCcnt DMA path
    load_tile_b(Bs, W2e, kk, n0, C_);
    if (kk + 32 < I_)
      __builtin_prefetch(W2e + (long)(kk + 32) * C_ + n0 + (tid & 7) * 8, 0, 0);
    wait_async_lds();
    __syncthreads();
    v16bf a0 = frag_A(As, wmb), a1 = frag_A(As, wmb + 1);
    v16bf b0 = frag_B(Bs, wnb), b1 = frag_B(Bs, wnb + 1);
    acc[0][0] = wmma_bf16(a0, b0, acc[0][0]);
    acc[0][1] = wmma_bf16(a0, b1, acc[0][1]);
    acc[1][0] = wmma_bf16(a1, b0, acc[1][0]);
    acc[1][1] = wmma_bf16(a1, b1, acc[1][1]);
  }
  const int hl = lane >> 4, c = lane & 15;
#pragma unroll
  for (int i = 0; i < 2; ++i) {
#pragma unroll
    for (int j = 0; j < 2; ++j) {
#pragma unroll
      for (int v = 0; v < 8; ++v) {
        long row = m0 + (wmb + i) * 16 + v + 8 * hl;
        long idx = row * C_ + n0 + (wnb + j) * 16 + c;
        out[idx] += acc[i][j][v];
      }
    }
  }
}

// ---------------------------------------------------------------- launch
extern "C" void kernel_launch(void* const* d_in, const int* in_sizes, int n_in,
                              void* d_out, int out_size, void* d_ws,
                              size_t ws_size, hipStream_t stream) {
  (void)in_sizes; (void)n_in; (void)out_size; (void)ws_size;
  const float* x        = (const float*)d_in[0];
  const float* ln1w     = (const float*)d_in[1];
  const float* ln1b     = (const float*)d_in[2];
  const float* ln2w     = (const float*)d_in[3];
  const float* ln2b     = (const float*)d_in[4];
  const float* ssim     = (const float*)d_in[5];
  const float* sgates   = (const float*)d_in[6];
  const float* q_proj   = (const float*)d_in[7];
  const float* k_proj   = (const float*)d_in[8];
  const float* v_proj   = (const float*)d_in[9];
  const float* o_proj   = (const float*)d_in[10];
  const float* msim     = (const float*)d_in[11];
  const float* mgates   = (const float*)d_in[12];
  const float* w1       = (const float*)d_in[13];
  const float* w2       = (const float*)d_in[14];
  float* out = (float*)d_out;

  size_t off = 0;
  auto take = [&](size_t bytes) -> char* {
    char* r = (char*)d_ws + off;
    off += (bytes + 255) & ~(size_t)255;
    return r;
  };
  bf16*  wqkv   = (bf16*)take((size_t)E_ * C_ * 3 * H_ * 2);
  bf16*  wo     = (bf16*)take((size_t)E_ * H_ * C_ * 2);
  bf16*  w1b    = (bf16*)take((size_t)E_ * C_ * I_ * 2);
  bf16*  w2b    = (bf16*)take((size_t)E_ * I_ * C_ * 2);
  bf16*  nbuf   = (bf16*)take((size_t)NTOK * C_ * 2);
  float* rw1    = (float*)take((size_t)NTOK * E_ * 4);
  float* rw2    = (float*)take((size_t)NTOK * E_ * 4);
  bf16*  qkvb   = (bf16*)take((size_t)NTOK * 3 * H_ * 2);
  float* scores = (float*)take((size_t)B_ * T_ * T_ * 4);
  bf16*  attn   = (bf16*)take((size_t)NTOK * H_ * 2);
  float* hs     = (float*)take((size_t)NTOK * C_ * 4);
  bf16*  inter  = (bf16*)take((size_t)NTOK * I_ * 2);

  // 1) weight conversion / packing to bf16
  {
    long n = (long)E_ * C_ * H_;
    k_pack_qkv<<<(unsigned)((n + 255) / 256), 256, 0, stream>>>(q_proj, k_proj,
                                                                v_proj, wqkv);
  }
  {
    long n = (long)E_ * H_ * C_;
    k_cvt_bf16<<<(unsigned)((n + 255) / 256), 256, 0, stream>>>(o_proj, wo, n);
  }
  {
    long n = (long)E_ * C_ * I_;
    k_cvt_bf16<<<(unsigned)((n + 255) / 256), 256, 0, stream>>>(w1, w1b, n);
    k_cvt_bf16<<<(unsigned)((n + 255) / 256), 256, 0, stream>>>(w2, w2b, n);
  }

  // 2) LN1 + attention-expert gating
  k_ln_gate<<<NTOK, 256, 0, stream>>>(x, ln1w, ln1b, ssim, sgates, nbuf, rw1,
                                      nullptr);

  // 3) Q|K|V all-expert projection with fused rw reduction -> bf16 (8192x384)
  k_gemm_rw<<<dim3(3 * H_ / 64, NTOK / 128), 256, 0, stream>>>(
      nbuf, C_, wqkv, C_, 3 * H_, rw1, nullptr, nullptr, qkvb, 3 * H_);

  // 4) causal scores, softmax, P@V
  k_attn_scores<<<dim3(T_ / 64, T_ / 128, B_), 256, 0, stream>>>(
      qkvb, scores, 0.0883883476483f /* 1/sqrt(128) */);
  k_softmax<<<NTOK, 256, 0, stream>>>(scores);
  k_gemm_pv<<<dim3(H_ / 64, T_ / 128, B_), 256, 0, stream>>>(scores, qkvb, attn);

  // 5) O-projection with fused rw reduction + residual -> hs (f32)
  k_gemm_rw<<<dim3(C_ / 64, NTOK / 128), 256, 0, stream>>>(
      attn, H_, wo, H_, C_, rw1, x, hs, nullptr, C_);

  // 6) LN2 + MoE gating; also seed d_out = hs
  k_ln_gate<<<NTOK, 256, 0, stream>>>(hs, ln2w, ln2b, msim, mgates, nbuf, rw2,
                                      out);

  // 7) MoE: per expert, gelu(n2 @ W1_e) * rw2 -> inter, then d_out += inter @ W2_e
  for (int e = 0; e < E_; ++e) {
    k_gemm_gelu<<<dim3(I_ / 64, NTOK / 128), 256, 0, stream>>>(
        nbuf, w1b + (size_t)e * C_ * I_, rw2, e, inter);
    k_gemm_acc<<<dim3(C_ / 64, NTOK / 128), 256, 0, stream>>>(
        inter, w2b + (size_t)e * I_ * C_, out);
  }
}
